// Gated_DERNN_GRU_TOPIC_59545426592304
// MI455X (gfx1250) — matrix operand order repaired and checked by reading, not compiled, not run on hardware
//
#include <hip/hip_runtime.h>
#include <hip/hip_bf16.h>

// ---------------- problem constants ----------------
#define HDIM   256
#define EDIM   300
#define EPAD   320            // E padded to multiple of 32 for K-loop
#define H3     768            // 3*H
#define NSENT  1024
#define LWORDS 64
#define NTOPIC 100
#define NTGT   5
#define MROWS  32             // sentences per tree block (2 M-tiles)

// ---------------- vector types ----------------
typedef __attribute__((ext_vector_type(16))) __bf16           v16bf;
typedef __attribute__((ext_vector_type(8)))  float            v8f;
typedef __attribute__((ext_vector_type(16))) unsigned short   v16u;
typedef __attribute__((ext_vector_type(8)))  unsigned short   v8u;
typedef __attribute__((ext_vector_type(4)))  unsigned short   v4u;

// ---------------- helpers ----------------
__device__ __forceinline__ unsigned short f2bf(float f) {
    unsigned int u = __float_as_uint(f);
    return (unsigned short)((u + 0x7FFFu + ((u >> 16) & 1u)) >> 16);   // RNE
}
__device__ __forceinline__ float bf2f(unsigned short h) {
    return __uint_as_float(((unsigned int)h) << 16);
}
__device__ __forceinline__ float sigf(float x) {
    return 1.0f / (1.0f + __expf(-x));
}

__device__ __forceinline__ v16bf make_v16bf(v8u lo, v8u hi) {
    v16u r;
#pragma unroll
    for (int i = 0; i < 8; ++i) { r[i] = lo[i]; r[i + 8] = hi[i]; }
    return __builtin_bit_cast(v16bf, r);
}

// A fragment (16x32 bf16, row M = lane%16):
//   lane<16: K = k0+{0..15};  lane>=16: K = k0+{16..31}
__device__ __forceinline__ v16bf load_a_u16(const unsigned short* base, int rowStride,
                                            int k0, int lane) {
    int m  = lane & 15;
    int kb = (lane < 16) ? 0 : 16;
    const unsigned short* p = base + m * rowStride + k0 + kb;
    v8u lo = *(const v8u*)(p);
    v8u hi = *(const v8u*)(p + 8);
    return make_v16bf(lo, hi);
}

// B fragment (32x16 bf16, column N = n0 + lane%16):
//   lane<16: K = k0+{0..15};  lane>=16: K = k0+{16..31}.  Weight stored row-major [N][K].
__device__ __forceinline__ v16bf load_b_u16(const unsigned short* base, int kStride,
                                            int n0, int k0, int lane) {
    int n  = n0 + (lane & 15);
    int ko = (lane < 16) ? 0 : 16;
    const unsigned short* p = base + (size_t)n * kStride + k0 + ko;
    v8u lo = *(const v8u*)(p);
    v8u hi = *(const v8u*)(p + 8);
    return make_v16bf(lo, hi);
}

// ---------------- workspace layout (bytes) ----------------
#define O_WBF   0u                      // [768][320] bf16  (W, zero-padded K)
#define O_UBF   491520u                 // [768][256] bf16  (U)
#define O_GWBF  884736u                 // [768][256] bf16  (gru_wih)
#define O_QD2   1277952u                // [10][768]  f32   (q @ D^T + b)
#define O_SVBF  1308672u                // [1024][256] bf16 (sentence vectors)
#define O_GI    1832960u                // [1024][768] f32  (gru input proj + bih)
#define O_HT    4978688u                // [256] f32        (final GRU hidden)

// ================================================================
// Prep 1: convert W (pad 300->320), U, gru_wih to bf16 row-major
// ================================================================
__global__ __launch_bounds__(256) void k_prep_conv(
    const float* __restrict__ W, const float* __restrict__ U,
    const float* __restrict__ gwih,
    unsigned short* __restrict__ Wbf, unsigned short* __restrict__ Ubf,
    unsigned short* __restrict__ Gbf)
{
    int i = blockIdx.x * 256 + threadIdx.x;
    if (i < H3 * EPAD) {
        int n = i / EPAD, e = i - n * EPAD;
        Wbf[i] = f2bf(e < EDIM ? W[n * EDIM + e] : 0.0f);
    }
    if (i < H3 * HDIM) {
        Ubf[i] = f2bf(U[i]);
        Gbf[i] = f2bf(gwih[i]);
    }
}

// ================================================================
// Prep 2: qD2[d][n] = sum_h D[n][h]*q[d][h] + b[n]   (10x768, tiny)
// ================================================================
__global__ __launch_bounds__(256) void k_prep_qd(
    const float* __restrict__ q, const float* __restrict__ D,
    const float* __restrict__ b, float* __restrict__ qD2)
{
    int i = blockIdx.x * 256 + threadIdx.x;
    if (i >= 10 * H3) return;
    int d = i / H3, n = i - d * H3;
    float a = b[n];
    const float* Dr = D + (size_t)n * HDIM;
    const float* qr = q + (size_t)d * HDIM;
#pragma unroll 8
    for (int h = 0; h < HDIM; ++h) a += Dr[h] * qr[h];
    qD2[i] = a;
}

// ================================================================
// Tree DERNN: one block = 32 sentences (2 M-tiles), full 64-step chain.
//   step 0  -> leaf node l=63 (dep = 9, no U@h / forget term)
//   step k  -> node l=63-k  : tot = x_l@W^T + h@U^T + qD2[dep]
//   h = tanh( sig(tot_i)*tanh(tot_u) + sig(tot_f)*h )
// 8 waves; each wave owns 2x6 accumulators (2 M-tiles x 96 columns).
// K outer / N inner: A fragments loaded once per K-tile, each streamed
// B tile feeds 2 WMMAs -> halves L2 weight traffic per sentence.
// ================================================================
__global__ __launch_bounds__(256) void k_tree(
    const int* __restrict__ sentIdx, const int* __restrict__ depTypes,
    const float* __restrict__ idx2vec,
    const unsigned short* __restrict__ Wbf, const unsigned short* __restrict__ Ubf,
    const float* __restrict__ qD2,
    unsigned short* __restrict__ SVbf)
{
    __shared__ __align__(16) unsigned short totbf[MROWS * H3];   // 48 KB
    __shared__ __align__(16) unsigned short xbf[MROWS * EPAD];   // 20 KB
    __shared__ __align__(16) unsigned short hbf[MROWS * HDIM];   // 16 KB (bf16 recurrent state)
    __shared__ int    widx[MROWS];
    __shared__ int    dep[MROWS];

    const int tid    = threadIdx.x;
    const int wave   = tid >> 5;
    const int lane   = tid & 31;
    const int sBase  = blockIdx.x * MROWS;
    const int n_base = wave * 96;          // this wave's first output column

    for (int step = 0; step < LWORDS; ++step) {
        const int  l    = (LWORDS - 1) - step;
        const bool leaf = (step == 0);

        if (tid < MROWS) {
            widx[tid] = sentIdx[(sBase + tid) * LWORDS + l];
            dep[tid]  = leaf ? 9 : depTypes[(sBase + tid) * (LWORDS - 1) + l];
        }
        __syncthreads();

        // gather this level's embeddings -> bf16 LDS (vectorized: 75 float4 = 300 f32)
        for (int i = tid; i < MROWS * 80; i += 256) {
            int m = i / 80, c = i - m * 80;
            v4u r = {0, 0, 0, 0};
            if (c < 75) {
                const float4 xx = *(const float4*)(idx2vec + (size_t)widx[m] * EDIM + c * 4);
                r[0] = f2bf(xx.x); r[1] = f2bf(xx.y); r[2] = f2bf(xx.z); r[3] = f2bf(xx.w);
            }
            *(v4u*)(xbf + m * EPAD + c * 4) = r;
        }
        __syncthreads();

        // ---- WMMA: tot[32][96 per wave] = x@W^T (+ h@U^T) + qD2[dep] ----
        {
            v8f acc[2][6];
#pragma unroll
            for (int mt = 0; mt < 2; ++mt) {
#pragma unroll
                for (int t6 = 0; t6 < 6; ++t6) {
                    int n = n_base + t6 * 16 + (lane & 15);
#pragma unroll
                    for (int j = 0; j < 8; ++j) {
                        int mj = (lane < 16) ? j : (j + 8);
                        acc[mt][t6][j] = qD2[dep[mt * 16 + mj] * H3 + n];
                    }
                }
            }
#pragma unroll 2
            for (int kt = 0; kt < EPAD / 32; ++kt) {
                v16bf a0 = load_a_u16(xbf,             EPAD, kt * 32, lane);
                v16bf a1 = load_a_u16(xbf + 16 * EPAD, EPAD, kt * 32, lane);
#pragma unroll
                for (int t6 = 0; t6 < 6; ++t6) {
                    v16bf bb = load_b_u16(Wbf, EPAD, n_base + t6 * 16, kt * 32, lane);
                    acc[0][t6] = __builtin_amdgcn_wmma_f32_16x16x32_bf16(
                                   false, a0, false, bb, (short)0, acc[0][t6], false, false);
                    acc[1][t6] = __builtin_amdgcn_wmma_f32_16x16x32_bf16(
                                   false, a1, false, bb, (short)0, acc[1][t6], false, false);
                }
            }
            if (!leaf) {
#pragma unroll 2
                for (int kt = 0; kt < HDIM / 32; ++kt) {
                    v16bf a0 = load_a_u16(hbf,             HDIM, kt * 32, lane);
                    v16bf a1 = load_a_u16(hbf + 16 * HDIM, HDIM, kt * 32, lane);
#pragma unroll
                    for (int t6 = 0; t6 < 6; ++t6) {
                        v16bf bb = load_b_u16(Ubf, HDIM, n_base + t6 * 16, kt * 32, lane);
                        acc[0][t6] = __builtin_amdgcn_wmma_f32_16x16x32_bf16(
                                       false, a0, false, bb, (short)0, acc[0][t6], false, false);
                        acc[1][t6] = __builtin_amdgcn_wmma_f32_16x16x32_bf16(
                                       false, a1, false, bb, (short)0, acc[1][t6], false, false);
                    }
                }
            }
#pragma unroll
            for (int mt = 0; mt < 2; ++mt) {
#pragma unroll
                for (int t6 = 0; t6 < 6; ++t6) {
                    int n = n_base + t6 * 16 + (lane & 15);
#pragma unroll
                    for (int j = 0; j < 8; ++j) {
                        int mj = (lane < 16) ? j : (j + 8);
                        totbf[(mt * 16 + mj) * H3 + n] = f2bf(acc[mt][t6][j]);
                    }
                }
            }
        }
        __syncthreads();

        // ---- pointwise gate update (bf16 state) ----
        for (int i = tid; i < MROWS * HDIM; i += 256) {
            int   m  = i >> 8, j = i & 255;
            float ti = bf2f(totbf[m * H3 + HDIM + j]);
            float tu = bf2f(totbf[m * H3 + 2 * HDIM + j]);
            float hn;
            if (leaf) {
                hn = tanhf(sigf(ti) * tanhf(tu));
            } else {
                float tf = bf2f(totbf[m * H3 + j]);
                hn = tanhf(sigf(ti) * tanhf(tu) + sigf(tf) * bf2f(hbf[i]));
            }
            hbf[i] = f2bf(hn);
        }
        __syncthreads();
    }

    // root hidden = sentence vector (bf16 for the GI GEMM)
    for (int i = tid; i < MROWS * HDIM; i += 256) {
        int m = i >> 8, j = i & 255;
        SVbf[(size_t)(sBase + m) * HDIM + j] = hbf[i];
    }
}

// ================================================================
// GI = SV @ gru_wih^T + gru_bih  : (1024 x 256) x (256 x 768) WMMA GEMM
// one wave per 16x16 tile; 64*48 = 3072 tiles; 8 waves/block -> 384 blocks
// ================================================================
__global__ __launch_bounds__(256) void k_gi_gemm(
    const unsigned short* __restrict__ SVbf, const unsigned short* __restrict__ Gbf,
    const float* __restrict__ bih, float* __restrict__ GI)
{
    const int lane = threadIdx.x & 31;
    const int wgid = blockIdx.x * 8 + (threadIdx.x >> 5);
    const int mt   = wgid / 48;
    const int nt   = wgid - mt * 48;
    const int n0   = nt * 16;
    const int n    = n0 + (lane & 15);

    v8f acc;
    {
        float bv = bih[n];
#pragma unroll
        for (int j = 0; j < 8; ++j) acc[j] = bv;
    }
    const unsigned short* Abase = SVbf + (size_t)mt * 16 * HDIM;
#pragma unroll 2
    for (int kt = 0; kt < HDIM / 32; ++kt) {
        v16bf a  = load_a_u16(Abase, HDIM, kt * 32, lane);
        v16bf bb = load_b_u16(Gbf, HDIM, n0, kt * 32, lane);
        acc = __builtin_amdgcn_wmma_f32_16x16x32_bf16(
                false, a, false, bb, (short)0, acc, false, false);
    }
#pragma unroll
    for (int j = 0; j < 8; ++j) {
        int mj = (lane < 16) ? j : (j + 8);
        GI[(size_t)(mt * 16 + mj) * H3 + n] = acc[j];
    }
}

// ================================================================
// Sequential GRU over 1024 sentence vectors (batch 1).
// One block, 768 threads: thread n owns row n of gru_whh matvec.
// ================================================================
__global__ __launch_bounds__(768) void k_gru(
    const float* __restrict__ GI, const float* __restrict__ whh,
    const float* __restrict__ bhh, const float* __restrict__ h0,
    float* __restrict__ hT)
{
    __shared__ float h[HDIM];
    __shared__ float gh[H3];
    const int tid = threadIdx.x;
    if (tid < HDIM) h[tid] = h0[tid];
    __syncthreads();

    const float* wr = whh + (size_t)tid * HDIM;
    const float  bv = bhh[tid];

    for (int t = 0; t < NSENT; ++t) {
        float acc = bv;
#pragma unroll 8
        for (int j = 0; j < HDIM; ++j) acc += wr[j] * h[j];
        gh[tid] = acc;
        __syncthreads();

        float hn = 0.0f;
        if (tid < HDIM) {
            const float* gi = GI + (size_t)t * H3;
            float r = sigf(gi[tid] + gh[tid]);
            float z = sigf(gi[HDIM + tid] + gh[HDIM + tid]);
            float nn = tanhf(gi[2 * HDIM + tid] + r * gh[2 * HDIM + tid]);
            hn = (1.0f - z) * nn + z * h[tid];
        }
        __syncthreads();
        if (tid < HDIM) h[tid] = hn;
        __syncthreads();
    }
    if (tid < HDIM) hT[tid] = h[tid];
}

// ================================================================
// Epilogue: topic MLP, gate fusion, classifier + softmax (1 block)
// ================================================================
__global__ __launch_bounds__(256) void k_final(
    const float* __restrict__ hTg, const float* __restrict__ DT,
    const float* __restrict__ mlpW, const float* __restrict__ mlpb,
    const float* __restrict__ gateW, const float* __restrict__ gateU,
    const float* __restrict__ gateb,
    const float* __restrict__ outW, const float* __restrict__ outb,
    float* __restrict__ out)
{
    __shared__ float hT[HDIM], tv[HDIM], g[2 * HDIM], v[HDIM], logits[NTGT];
    const int tid = threadIdx.x;

    hT[tid] = hTg[tid];
    {
        float a = mlpb[tid];
        const float* mr = mlpW + (size_t)tid * NTOPIC;
#pragma unroll 4
        for (int c = 0; c < NTOPIC; ++c) a += mr[c] * DT[c];
        tv[tid] = tanhf(a);
    }
    __syncthreads();

    for (int i = tid; i < 2 * HDIM; i += 256) {
        float a = gateb[i];
        const float* gw = gateW + (size_t)i * HDIM;
        const float* gu = gateU + (size_t)i * HDIM;
#pragma unroll 8
        for (int j = 0; j < HDIM; ++j) a += gw[j] * hT[j] + gu[j] * tv[j];
        g[i] = a;
    }
    __syncthreads();

    v[tid] = tanhf(sigf(g[tid]) * hT[tid] + sigf(g[HDIM + tid]) * tv[tid]);
    __syncthreads();

    if (tid < NTGT) {
        float a = outb[tid];
        const float* ow = outW + (size_t)tid * HDIM;
#pragma unroll 8
        for (int j = 0; j < HDIM; ++j) a += ow[j] * v[j];
        logits[tid] = a;
    }
    __syncthreads();

    if (tid == 0) {
        float mx = logits[0];
#pragma unroll
        for (int i = 1; i < NTGT; ++i) mx = fmaxf(mx, logits[i]);
        float e[NTGT], s = 0.0f;
#pragma unroll
        for (int i = 0; i < NTGT; ++i) { e[i] = __expf(logits[i] - mx); s += e[i]; }
#pragma unroll
        for (int i = 0; i < NTGT; ++i) out[i] = e[i] / s;
    }
}

// ================================================================
extern "C" void kernel_launch(void* const* d_in, const int* in_sizes, int n_in,
                              void* d_out, int out_size, void* d_ws, size_t ws_size,
                              hipStream_t stream) {
    const int*   sentIdx = (const int*)  d_in[0];
    const int*   depT    = (const int*)  d_in[1];
    const float* DT      = (const float*)d_in[2];
    const float* h0      = (const float*)d_in[3];
    const float* idx2vec = (const float*)d_in[4];
    const float* q       = (const float*)d_in[5];
    const float* W       = (const float*)d_in[6];
    const float* U       = (const float*)d_in[7];
    const float* D       = (const float*)d_in[8];
    const float* b       = (const float*)d_in[9];
    const float* gwih    = (const float*)d_in[10];
    const float* gwhh    = (const float*)d_in[11];
    const float* gbih    = (const float*)d_in[12];
    const float* gbhh    = (const float*)d_in[13];
    const float* gateW   = (const float*)d_in[14];
    const float* gateU   = (const float*)d_in[15];
    const float* gateb   = (const float*)d_in[16];
    const float* mlpW    = (const float*)d_in[17];
    const float* mlpb    = (const float*)d_in[18];
    const float* outW    = (const float*)d_in[19];
    const float* outb    = (const float*)d_in[20];
    float* out = (float*)d_out;
    char*  ws  = (char*)d_ws;

    unsigned short* Wbf  = (unsigned short*)(ws + O_WBF);
    unsigned short* Ubf  = (unsigned short*)(ws + O_UBF);
    unsigned short* Gbf  = (unsigned short*)(ws + O_GWBF);
    float*          qD2  = (float*)         (ws + O_QD2);
    unsigned short* SVbf = (unsigned short*)(ws + O_SVBF);
    float*          GI   = (float*)         (ws + O_GI);
    float*          hT   = (float*)         (ws + O_HT);

    k_prep_conv<<<(H3 * EPAD + 255) / 256, 256, 0, stream>>>(W, U, gwih, Wbf, Ubf, Gbf);
    k_prep_qd  <<<(10 * H3 + 255) / 256,  256, 0, stream>>>(q, D, b, qD2);
    k_tree     <<<NSENT / MROWS,          256, 0, stream>>>(sentIdx, depT, idx2vec,
                                                            Wbf, Ubf, qD2, SVbf);
    k_gi_gemm  <<<(NSENT / 16) * 48 / 8,  256, 0, stream>>>(SVbf, Gbf, gbih, GI);
    k_gru      <<<1,                      768, 0, stream>>>(GI, gwhh, gbhh, h0, hT);
    k_final    <<<1,                      256, 0, stream>>>(hT, DT, mlpW, mlpb,
                                                            gateW, gateU, gateb,
                                                            outW, outb, out);
}
